// CorrLSTM_upsample_resize_PSLA_29609504539001
// MI455X (gfx1250) — matrix operand order repaired and checked by reading, not compiled, not run on hardware
//
#include <hip/hip_runtime.h>
#include <hip/hip_bf16.h>
#include <math.h>

typedef __attribute__((ext_vector_type(16))) __bf16 v16bf;
typedef __attribute__((ext_vector_type(8)))  __bf16 v8bf;
typedef __attribute__((ext_vector_type(8)))  float  v8f;

#define NB 4
#define PPC 81

union V16 { v16bf v; v8bf h[2]; };

__device__ __forceinline__ int fdiv_floor(int a, int f) {
  return (a >= 0) ? (a / f) : -((-a + f - 1) / f);
}

// ---- zero the pad border of a padded NHWC bf16 buffer -----------------------
__global__ void zero_border(__bf16* __restrict__ dst, int C, int s, int sp,
                            int pad, int total) {
  int t = blockIdx.x * blockDim.x + threadIdx.x;
  if (t >= total) return;
  int rest = t / C;
  int xq = rest % sp; rest /= sp;
  int yq = rest % sp;
  int x = xq - pad, y = yq - pad;
  if (x >= 0 && x < s && y >= 0 && y < s) return;   // interior: untouched
  dst[t] = (__bf16)0.f;
}

// ---- fallback pack (small levels): NCHW f32 -> padded NHWC bf16 -------------
__global__ void pack_feat_pad(const float* __restrict__ src, __bf16* __restrict__ dst,
                              int C, int s, int sp, int total) {
  int t = blockIdx.x * blockDim.x + threadIdx.x;
  if (t >= total) return;
  int c = t % C; int rest = t / C;
  int xq = rest % sp; rest /= sp;
  int yq = rest % sp; int b = rest / sp;
  int x = xq - 4, y = yq - 4;
  float v = 0.f;
  if (x >= 0 && x < s && y >= 0 && y < s)
    v = src[(((size_t)b * C + c) * s + y) * s + x];
  dst[t] = (__bf16)v;
}

// ---- coalesced tiled-transpose pack: NCHW f32 -> padded NHWC bf16 -----------
// 64 spatial x 64 channel tile via LDS; requires s*s % 64 == 0, C % 64 == 0.
__global__ __launch_bounds__(256) void pack_feat_t(
    const float* __restrict__ src, __bf16* __restrict__ dst,
    int C, int s, int sp) {
  __shared__ float tile[64][65];
  const int b = blockIdx.z;
  const int c0 = blockIdx.y * 64;
  const int p0 = blockIdx.x * 64;
  const int ss = s * s;
  for (int i = threadIdx.x; i < 64 * 64; i += 256) {
    int cr = i >> 6, pc = i & 63;                  // consecutive pc: coalesced
    tile[cr][pc] = src[((size_t)b * C + c0 + cr) * ss + p0 + pc];
  }
  __syncthreads();
  for (int i = threadIdx.x; i < 64 * 64; i += 256) {
    int pr = i >> 6, cc = i & 63;                  // consecutive cc: coalesced
    int p = p0 + pr, y = p / s, x = p % s;
    dst[(size_t)((y + 4) * sp + (x + 4)) * C + c0 + cc] = (__bf16)tile[cc][pr];
  }
}

// ---- pack conv weights into WMMA A-fragment layout --------------------------
// dst[tap][mtile(21)][mlane(16)][k(192)] ; oc = mtile*16+mlane, k = in-channel
__global__ void pack_w(const float* __restrict__ w, __bf16* __restrict__ dst, int total) {
  int t = blockIdx.x * blockDim.x + threadIdx.x;
  if (t >= total) return;
  int k = t % 192; int rest = t / 192;
  int ml = rest % 16; rest /= 16;
  int mt = rest % 21; int tap = rest / 21;
  int oc = mt * 16 + ml;
  float v = 0.f;
  if (oc < 324 && k < 162) v = w[((size_t)oc * 162 + k) * 9 + tap];
  dst[t] = (__bf16)v;
}

__device__ __forceinline__ float xin_src(const float* dw, const float* h,
                                         int b, int c, int ss, int p) {
  if (c < 81)  return dw[((size_t)b * PPC + c) * ss + p];
  if (c < 162) return h[((size_t)b * PPC + (c - 81)) * ss + p];
  return 0.f;
}

// ---- fallback x_in pack (small levels) --------------------------------------
__global__ void pack_xin_pad(const float* __restrict__ dw, const float* __restrict__ h,
                             __bf16* __restrict__ dst, int s, int sq, int total) {
  int t = blockIdx.x * blockDim.x + threadIdx.x;
  if (t >= total) return;
  int c = t % 192; int rest = t / 192;
  int xq = rest % sq; rest /= sq;
  int yq = rest % sq; int b = rest / sq;
  int x = xq - 1, y = yq - 1;
  float v = 0.f;
  if (x >= 0 && x < s && y >= 0 && y < s)
    v = xin_src(dw, h, b, c, s * s, y * s + x);
  dst[t] = (__bf16)v;
}

// ---- coalesced tiled-transpose x_in pack ------------------------------------
__global__ __launch_bounds__(256) void pack_xin_t(
    const float* __restrict__ dw, const float* __restrict__ h,
    __bf16* __restrict__ dst, int s, int sq) {
  __shared__ float tile[64][65];
  const int b = blockIdx.z;
  const int c0 = blockIdx.y * 64;
  const int p0 = blockIdx.x * 64;
  const int ss = s * s;
  for (int i = threadIdx.x; i < 64 * 64; i += 256) {
    int cr = i >> 6, pc = i & 63;
    tile[cr][pc] = xin_src(dw, h, b, c0 + cr, ss, p0 + pc);
  }
  __syncthreads();
  for (int i = threadIdx.x; i < 64 * 64; i += 256) {
    int pr = i >> 6, cc = i & 63;
    int p = p0 + pr, y = p / s, x = p % s;
    dst[(size_t)((y + 1) * sq + (x + 1)) * 192 + c0 + cc] = (__bf16)tile[cc][pr];
  }
}

// ---- correlation cost volume via bf16 WMMA ----------------------------------
// One wave per 4x4 tile. Gram: 144 window rows (9 M-tiles) x 16 positions,
// K = C in chunks of 32 bf16. Padded inputs -> branch-free loads.
__global__ __launch_bounds__(32) void corr_wmma(
    const __bf16* __restrict__ f1p, const __bf16* __restrict__ f2p,
    float* __restrict__ dw, int C, int s, int sp, int f) {
  __shared__ float G[144][17];
  const int lane = threadIdx.x;
  const int l15 = lane & 15, lhi = lane >> 4;
  const int b = blockIdx.z;
  const int y0 = blockIdx.y * 4, x0 = blockIdx.x * 4;
  const __bf16* b1 = f1p + (size_t)b * sp * sp * C;
  const __bf16* b2 = f2p + (size_t)b * sp * sp * C;

  const int py = l15 >> 2, px = l15 & 3;
  const __bf16* bbase = b1 + (size_t)((y0 + py + 4) * sp + (x0 + px + 4)) * C + 8 * lhi;

  const __bf16* abase[9];
#pragma unroll
  for (int mt = 0; mt < 9; ++mt) {
    int m = mt * 16 + l15;
    int wy = m / 12, wx = m % 12;
    abase[mt] = b2 + (size_t)((y0 + wy) * sp + (x0 + wx)) * C + 8 * lhi;
  }

  v8f zf = {};
  v8f acc[9];
#pragma unroll
  for (int mt = 0; mt < 9; ++mt) acc[mt] = zf;

  for (int kc = 0; kc < C; kc += 32) {
    V16 bf;
    bf.h[0] = *(const v8bf*)(bbase + kc);
    bf.h[1] = *(const v8bf*)(bbase + kc + 16);
    V16 af[9];
#pragma unroll
    for (int mt = 0; mt < 9; ++mt) {
      af[mt].h[0] = *(const v8bf*)(abase[mt] + kc);
      af[mt].h[1] = *(const v8bf*)(abase[mt] + kc + 16);
    }
#pragma unroll
    for (int mt = 0; mt < 9; ++mt)
      acc[mt] = __builtin_amdgcn_wmma_f32_16x16x32_bf16(
          false, af[mt].v, false, bf.v, (short)0, acc[mt], false, false);
  }

#pragma unroll
  for (int mt = 0; mt < 9; ++mt)
#pragma unroll
    for (int r = 0; r < 8; ++r)
      G[mt * 16 + r + 8 * lhi][l15] = acc[mt][r];
  __syncthreads();

  for (int t = lane; t < PPC * 16; t += 32) {
    int idx = t >> 4, n = t & 15;
    int i = idx / 9, j = idx % 9;
    int dy = fdiv_floor(i - 4, f), dx = fdiv_floor(j - 4, f);
    int opy = n >> 2, opx = n & 3;
    int y = y0 + opy, x = x0 + opx;
    if (y >= s || x >= s) continue;
    int ya = y + dy, xa = x + dx;
    float v = 0.f;
    if (ya >= 0 && ya < s && xa >= 0 && xa < s)
      v = G[(opy + dy + 4) * 12 + (opx + dx + 4)][n];
    v = (v >= 0.f) ? v : 0.01f * v;
    dw[(((size_t)b * PPC + idx) * s + y) * s + x] = v;
  }
}

// ---- 3x3 conv (162->324 ch) implicit GEMM via bf16 WMMA ---------------------
// 256 threads / 8 waves per block covering 8x8 spatial (4 sub-tiles): each
// weight A-fragment loaded from global is reused by 4 WMMAs -> 4x less L2
// weight traffic than 4x4 blocks. 10x10x192 bf16 patch staged in LDS.
__global__ __launch_bounds__(256) void conv_wmma(
    const __bf16* __restrict__ xin, const __bf16* __restrict__ wbf,
    float* __restrict__ cw, int s, int sq) {
  __shared__ __bf16 patch[10 * 10 * 192];   // 38.4 KB
  const int b = blockIdx.z;
  const int Y0 = blockIdx.y * 8, X0 = blockIdx.x * 8;
  const __bf16* xb = xin + (size_t)b * sq * sq * 192;

  for (int i = threadIdx.x; i < 2400; i += 256) {
    int cell = i / 24; int ck = (i % 24) * 8;
    int ry = cell / 10, rx = cell % 10;
    // interior (Y0-1+ry, X0-1+rx) == padded (Y0+ry, X0+rx): always in-bounds
    *(v8bf*)&patch[cell * 192 + ck] =
        *(const v8bf*)(xb + (size_t)((Y0 + ry) * sq + (X0 + rx)) * 192 + ck);
  }
  __syncthreads();

  const int wave = threadIdx.x >> 5, lane = threadIdx.x & 31;
  const int l15 = lane & 15, lhi = lane >> 4;
  const int py = l15 >> 2, px = l15 & 3;

  for (int mt = wave; mt < 21; mt += 8) {
    __builtin_prefetch(wbf + (size_t)(mt * 16 + l15) * 192, 0, 1);
    v8f acc[4];
#pragma unroll
    for (int st = 0; st < 4; ++st) acc[st] = v8f{};
#pragma unroll
    for (int tap = 0; tap < 9; ++tap) {
      int ky = tap / 3, kx = tap % 3;
      const __bf16* wp = wbf + (size_t)((tap * 21 + mt) * 16 + l15) * 192 + 8 * lhi;
      for (int kc = 0; kc < 192; kc += 32) {
        V16 af;
        af.h[0] = *(const v8bf*)(wp + kc);
        af.h[1] = *(const v8bf*)(wp + kc + 16);
#pragma unroll
        for (int st = 0; st < 4; ++st) {
          int sy = (st >> 1) * 4, sx = (st & 1) * 4;
          const __bf16* lp =
              &patch[((sy + py + ky) * 10 + (sx + px + kx)) * 192 + kc + 8 * lhi];
          V16 bf;
          bf.h[0] = *(const v8bf*)(lp);
          bf.h[1] = *(const v8bf*)(lp + 16);
          acc[st] = __builtin_amdgcn_wmma_f32_16x16x32_bf16(
              false, af.v, false, bf.v, (short)0, acc[st], false, false);
        }
      }
    }
#pragma unroll
    for (int st = 0; st < 4; ++st) {
      int oy = Y0 + (st >> 1) * 4 + py, ox = X0 + (st & 1) * 4 + px;
      if (oy < s && ox < s) {
#pragma unroll
        for (int r = 0; r < 8; ++r) {
          int oc = mt * 16 + r + 8 * lhi;
          if (oc < 324)
            cw[(((size_t)b * 324 + oc) * s + oy) * s + ox] = acc[st][r];
        }
      }
    }
  }
}

// ---- LSTM gates (elementwise) -----------------------------------------------
__global__ void lstm_gates(const float* __restrict__ cw, const float* __restrict__ bias,
                           const float* __restrict__ cpre, float* __restrict__ hout,
                           float* __restrict__ cout_, int s, int total) {
  int t = blockIdx.x * blockDim.x + threadIdx.x;
  if (t >= total) return;
  int ss = s * s;
  int yx = t % ss; int rest = t / ss;
  int ch = rest % PPC; int b = rest / PPC;
  size_t base = (size_t)b * 324 * ss + yx;
  float ci = cw[base + (size_t)(ch)*ss] + bias[ch];
  float cf = cw[base + (size_t)(ch + 81) * ss] + bias[ch + 81];
  float co = cw[base + (size_t)(ch + 162) * ss] + bias[ch + 162];
  float cg = cw[base + (size_t)(ch + 243) * ss] + bias[ch + 243];
  float ig = 1.f / (1.f + __expf(-ci));
  float fg = 1.f / (1.f + __expf(-cf));
  float og = 1.f / (1.f + __expf(-co));
  float g = tanhf(cg);
  float cn = fg * cpre[t] + ig * g;
  hout[t] = og * tanhf(cn);
  cout_[t] = cn;
}

// ---- new_feat: softmax(81) + weighted 9x9 window sum (padded f1) ------------
__global__ __launch_bounds__(256) void new_feat(const float* __restrict__ dw,
                                                const __bf16* __restrict__ f1p,
                                                float* __restrict__ nf,
                                                int C, int s, int sp) {
  __shared__ float sdw[PPC];
  __shared__ float sprob[PPC];
  __shared__ float sred[2];
  const int b = blockIdx.y;
  const int pos = blockIdx.x;
  const int y = pos / s, x = pos % s;
  const int t = threadIdx.x;
  const int ss = s * s;
  if (t < PPC) sdw[t] = dw[((size_t)b * PPC + t) * ss + pos];
  __syncthreads();
  if (t == 0) {
    float mx = sdw[0];
    for (int k = 1; k < PPC; ++k) mx = fmaxf(mx, sdw[k]);
    float sum = 0.f;
    for (int k = 0; k < PPC; ++k) sum += __expf(sdw[k] - mx);
    sred[0] = mx; sred[1] = 1.f / sum;
  }
  __syncthreads();
  if (t < PPC) sprob[t] = __expf(sdw[t] - sred[0]) * sred[1];
  __syncthreads();
  const __bf16* fb = f1p + (size_t)b * sp * sp * C;
  for (int c = t; c < C; c += 256) {
    float acc = 0.f;
    for (int idx = 0; idx < PPC; ++idx) {
      int dy = idx / 9 - 4, dx = idx % 9 - 4;
      acc += sprob[idx] * (float)fb[(size_t)((y + dy + 4) * sp + (x + dx + 4)) * C + c];
    }
    nf[((size_t)b * C + c) * ss + pos] = acc;
  }
}

// ---- host launcher ----------------------------------------------------------
extern "C" void kernel_launch(void* const* d_in, const int* in_sizes, int n_in,
                              void* d_out, int out_size, void* d_ws, size_t ws_size,
                              hipStream_t stream) {
  static const int CHs[6] = {512, 1024, 512, 256, 256, 256};
  static const int SPs[6] = {64, 32, 16, 8, 4, 2};

  char* ws = (char*)d_ws;
  size_t off = 0;
  auto alloc = [&](size_t bytes) {
    size_t o = off;
    off = (off + bytes + 255) & ~(size_t)255;
    return o;
  };

  size_t xbf_o[6], xpbf_o[6], dw_o[6], xin_o[6], wbf_o[6], cw_o[6];
  for (int i = 0; i < 6; ++i) {
    int s = SPs[i];
    size_t sp = 4 * ((s + 3) / 4) + 8;       // corr pad (4-px border)
    size_t sq = 8 * ((s + 7) / 8) + 2;       // conv pad (1-px border, 8x8 tiles)
    size_t ss = (size_t)s * s;
    xbf_o[i]  = alloc((size_t)NB * sp * sp * CHs[i] * 2);
    xpbf_o[i] = alloc((size_t)NB * sp * sp * CHs[i] * 2);
    dw_o[i]   = alloc((size_t)NB * PPC * ss * 4);
    xin_o[i]  = alloc((size_t)NB * sq * sq * 192 * 2);
    wbf_o[i]  = alloc((size_t)9 * 21 * 16 * 192 * 2);
    cw_o[i]   = alloc((size_t)NB * 324 * ss * 4);
  }

  float* out = (float*)d_out;
  size_t hoff[6], coff[6], nfoff[6], ooff = 0;
  for (int i = 0; i < 6; ++i) {
    size_t ss = (size_t)SPs[i] * SPs[i];
    hoff[i] = ooff; ooff += (size_t)NB * PPC * ss;
    coff[i] = ooff; ooff += (size_t)NB * PPC * ss;
  }
  for (int i = 0; i < 6; ++i) {
    size_t ss = (size_t)SPs[i] * SPs[i];
    nfoff[i] = ooff; ooff += (size_t)NB * CHs[i] * ss;
  }

  for (int i = 0; i < 6; ++i) {
    const int C = CHs[i], s = SPs[i], f = 1 << i;
    const int ss = s * s;
    const int sp = 4 * ((s + 3) / 4) + 8;
    const int sq = 8 * ((s + 7) / 8) + 2;
    const float* x  = (const float*)d_in[6 * i + 0];
    const float* xp = (const float*)d_in[6 * i + 1];
    const float* h  = (const float*)d_in[6 * i + 2];
    const float* cp = (const float*)d_in[6 * i + 3];
    const float* w  = (const float*)d_in[6 * i + 4];
    const float* bs = (const float*)d_in[6 * i + 5];
    __bf16* xbf  = (__bf16*)(ws + xbf_o[i]);
    __bf16* xpbf = (__bf16*)(ws + xpbf_o[i]);
    float*  dwp  = (float*)(ws + dw_o[i]);
    __bf16* xin  = (__bf16*)(ws + xin_o[i]);
    __bf16* wbf  = (__bf16*)(ws + wbf_o[i]);
    float*  cwp  = (float*)(ws + cw_o[i]);

    if (ss % 64 == 0) {
      int totb = NB * sp * sp * C;
      zero_border<<<(totb + 255) / 256, 256, 0, stream>>>(xbf, C, s, sp, 4, totb);
      zero_border<<<(totb + 255) / 256, 256, 0, stream>>>(xpbf, C, s, sp, 4, totb);
      dim3 pg(ss / 64, C / 64, NB);
      pack_feat_t<<<pg, 256, 0, stream>>>(x, xbf, C, s, sp);
      pack_feat_t<<<pg, 256, 0, stream>>>(xp, xpbf, C, s, sp);
    } else {
      int tot1 = NB * sp * sp * C;
      pack_feat_pad<<<(tot1 + 255) / 256, 256, 0, stream>>>(x, xbf, C, s, sp, tot1);
      pack_feat_pad<<<(tot1 + 255) / 256, 256, 0, stream>>>(xp, xpbf, C, s, sp, tot1);
    }
    int totw = 9 * 21 * 16 * 192;
    pack_w<<<(totw + 255) / 256, 256, 0, stream>>>(w, wbf, totw);

    dim3 tg4((s + 3) / 4, (s + 3) / 4, NB);
    corr_wmma<<<tg4, 32, 0, stream>>>(xbf, xpbf, dwp, C, s, sp, f);

    if (ss % 64 == 0) {
      int totb = NB * sq * sq * 192;
      zero_border<<<(totb + 255) / 256, 256, 0, stream>>>(xin, 192, s, sq, 1, totb);
      dim3 pg(ss / 64, 3, NB);
      pack_xin_t<<<pg, 256, 0, stream>>>(dwp, h, xin, s, sq);
    } else {
      int totxi = NB * sq * sq * 192;
      pack_xin_pad<<<(totxi + 255) / 256, 256, 0, stream>>>(dwp, h, xin, s, sq, totxi);
    }

    dim3 tg8((s + 7) / 8, (s + 7) / 8, NB);
    conv_wmma<<<tg8, 256, 0, stream>>>(xin, wbf, cwp, s, sq);

    int totg = NB * PPC * ss;
    lstm_gates<<<(totg + 255) / 256, 256, 0, stream>>>(
        cwp, bs, cp, out + hoff[i], out + coff[i], s, totg);

    new_feat<<<dim3(ss, NB), 256, 0, stream>>>(dwp, xbf, out + nfoff[i], C, s, sp);
  }
  (void)in_sizes; (void)n_in; (void)out_size; (void)ws_size;
}
